// Qwen3MoeDecoderLayer_74457553043827
// MI455X (gfx1250) — matrix operand order repaired and checked by reading, not compiled, not run on hardware
//
#include <hip/hip_runtime.h>
#include <hip/hip_bf16.h>

// ---------------------------------------------------------------------------
// Qwen3-MoE decoder layer for gfx1250 (MI455X). Wave32, WMMA f16 (16x16x32).
// B=2 S=1024 H=2048 NH=16 NKV=4 HD=128 E=16 topK=4 I=768
// Fragment loads vectorized, contiguous k order (k = 16*hi + s) in A and B:
//   A 16x32 f16 : lane -> one contiguous 16-half run in its row  (2x b128)
//   B 32x16 f16 : lane -> one contiguous 16-half run over K      (1x 32B)
// B/V panels staged N-major; transpose packing done in registers so every
// LDS store is a 16-byte v8h store (no scalar ds_store_b16).
// ---------------------------------------------------------------------------

typedef __attribute__((ext_vector_type(16))) _Float16 v16h;
typedef __attribute__((ext_vector_type(8)))  _Float16 v8h;
typedef __attribute__((ext_vector_type(8)))  float    v8f;

constexpr int cB    = 2;
constexpr int cS    = 1024;
constexpr int cH    = 2048;
constexpr int cNH   = 16;
constexpr int cNKV  = 4;
constexpr int cHD   = 128;
constexpr int cE    = 16;
constexpr int cTOPK = 4;
constexpr int cI    = 768;
constexpr int cT    = cB * cS;            // 2048 tokens

// A-fragment: p points at the lane's row (32 halves). Two 16B loads + concat (k = 16*hi + s).
__device__ __forceinline__ v16h ld_frag_row(const _Float16* p, int hi) {
  v8h lo = *(const v8h*)(p + (hi ? 16 : 0));
  v8h hh = *(const v8h*)(p + 8 + (hi ? 16 : 0));
  return __builtin_shufflevector(lo, hh, 0, 1, 2, 3, 4, 5, 6, 7,
                                 8, 9, 10, 11, 12, 13, 14, 15);
}
// B-fragment: p points at a 32-half contiguous K-run for the lane's column.
__device__ __forceinline__ v16h ld_frag_contig(const _Float16* p, int hi) {
  return *(const v16h*)(p + (hi ? 16 : 0));
}

// ---------------- Generic WMMA GEMM: C[M,N] = A[M,K] * W[K,N] ----------------
// mode 0: C = acc ; mode 1: C = resid + acc ; mode 2: C += rowscale[row]*acc
#define GTM 128
#define GTN 64
#define GTK 32

__global__ __launch_bounds__(256)
void gemm_f16_wmma(const float* __restrict__ A, const float* __restrict__ W,
                   float* __restrict__ C, const float* __restrict__ resid,
                   const float* __restrict__ rowscale,
                   int M, int N, int K, int mode)
{
  __shared__ alignas(64) _Float16 As[GTM][GTK];   // 8 KB, row-major
  __shared__ alignas(64) _Float16 Bt[GTN][GTK];   // 4 KB, N-major (transposed)

  const int tid  = threadIdx.x;
  const int lane = tid & 31;
  const int wave = tid >> 5;
  const int hi   = (lane >= 16) ? 1 : 0;
  const int ln   = lane & 15;
  const int wm   = wave >> 1;        // 0..3 : 32-row slab
  const int wn   = wave & 1;         // 0..1 : 32-col slab
  const int m0   = blockIdx.y * GTM;
  const int n0   = blockIdx.x * GTN;

  // staging maps
  const int ar  = tid >> 1;          // 0..127 : A row
  const int ac  = (tid & 1) * 16;    // 0 / 16 : A col base (16 floats)
  const int bn2 = tid & 63;          // 0..63  : B column within tile
  const int bko = (tid >> 6) * 8;    // 0,8,16,24 : B k-octet base

  v8f c00 = {}, c01 = {}, c10 = {}, c11 = {};

  for (int kt = 0; kt < K; kt += GTK) {
    // A tile 128x32: contiguous 16-float runs, packed to f16, b128 LDS stores
    const float4* ap = (const float4*)(A + (size_t)(m0 + ar) * K + kt + ac);
#pragma unroll
    for (int i = 0; i < 4; ++i) {
      float4 f = ap[i];
      As[ar][ac + 4 * i + 0] = (_Float16)f.x;
      As[ar][ac + 4 * i + 1] = (_Float16)f.y;
      As[ar][ac + 4 * i + 2] = (_Float16)f.z;
      As[ar][ac + 4 * i + 3] = (_Float16)f.w;
    }
    // B tile 32x64 -> Bt[n][k]: thread packs an 8-long K-run for one column.
    // Per j, lanes read consecutive n -> coalesced b32 global loads.
    {
      union { _Float16 h[8]; v8h v; } pk;
#pragma unroll
      for (int j = 0; j < 8; ++j)
        pk.h[j] = (_Float16)W[(size_t)(kt + bko + j) * N + n0 + bn2];
      *(v8h*)&Bt[bn2][bko] = pk.v;
    }
    if (kt + GTK < K) {   // prefetch next k-tile (global_prefetch_b8)
      __builtin_prefetch(A + (size_t)(m0 + ar) * K + kt + GTK + ac, 0, 1);
      __builtin_prefetch(W + (size_t)(kt + GTK + bko) * N + n0 + bn2, 0, 1);
    }
    __syncthreads();

    v16h a0 = ld_frag_row(&As[wm * 32 + ln][0], hi);
    v16h a1 = ld_frag_row(&As[wm * 32 + 16 + ln][0], hi);
    v16h b0 = ld_frag_contig(&Bt[wn * 32 + ln][0], hi);
    v16h b1 = ld_frag_contig(&Bt[wn * 32 + 16 + ln][0], hi);
    c00 = __builtin_amdgcn_wmma_f32_16x16x32_f16(false, a0, false, b0, (short)0, c00, false, false);
    c01 = __builtin_amdgcn_wmma_f32_16x16x32_f16(false, a0, false, b1, (short)0, c01, false, false);
    c10 = __builtin_amdgcn_wmma_f32_16x16x32_f16(false, a1, false, b0, (short)0, c10, false, false);
    c11 = __builtin_amdgcn_wmma_f32_16x16x32_f16(false, a1, false, b1, (short)0, c11, false, false);
    __syncthreads();
  }

#pragma unroll
  for (int r = 0; r < 8; ++r) {
    int rr   = hi ? r + 8 : r;
    int row0 = m0 + wm * 32 + rr;
    int row1 = row0 + 16;
    int col0 = n0 + wn * 32 + ln;
    size_t i00 = (size_t)row0 * N + col0, i01 = i00 + 16;
    size_t i10 = (size_t)row1 * N + col0, i11 = i10 + 16;
    if (mode == 1) {
      C[i00] = resid[i00] + c00[r]; C[i01] = resid[i01] + c01[r];
      C[i10] = resid[i10] + c10[r]; C[i11] = resid[i11] + c11[r];
    } else if (mode == 2) {
      float s0 = rowscale[row0], s1 = rowscale[row1];
      C[i00] += s0 * c00[r]; C[i01] += s0 * c01[r];
      C[i10] += s1 * c10[r]; C[i11] += s1 * c11[r];
    } else {
      C[i00] = c00[r]; C[i01] = c01[r];
      C[i10] = c10[r]; C[i11] = c11[r];
    }
  }
}

// ---------------- Token RMSNorm over H=2048 ----------------
__global__ __launch_bounds__(256)
void rmsnorm_kernel(const float* __restrict__ x, const float* __restrict__ w,
                    float* __restrict__ y, int Hdim)
{
  __shared__ float red[8];
  const int t = blockIdx.x;
  const float* xr = x + (size_t)t * Hdim;
  float ss = 0.f;
  for (int i = threadIdx.x; i < Hdim; i += 256) { float v = xr[i]; ss += v * v; }
  for (int off = 1; off < 32; off <<= 1) ss += __shfl_xor(ss, off, 32);
  int wave = threadIdx.x >> 5, lane = threadIdx.x & 31;
  if (lane == 0) red[wave] = ss;
  __syncthreads();
  if (wave == 0) {
    float v = (lane < 8) ? red[lane] : 0.f;
    for (int off = 1; off < 8; off <<= 1) v += __shfl_xor(v, off, 32);
    if (lane == 0) red[0] = v;
  }
  __syncthreads();
  float inv = rsqrtf(red[0] / (float)Hdim + 1e-6f);
  for (int i = threadIdx.x; i < Hdim; i += 256)
    y[(size_t)t * Hdim + i] = xr[i] * inv * w[i];
}

// ---------------- Per-head QK RMSNorm + RoPE (one wave / 128-dim row) -------
__global__ __launch_bounds__(32)
void qknorm_rope_kernel(float* __restrict__ x, const float* __restrict__ w, int nh)
{
  const int row = blockIdx.x;            // (b*S + s)*nh + head
  const int s   = (row / nh) % cS;
  float* xr = x + (size_t)row * cHD;
  const int l = threadIdx.x;

  float x0 = xr[l], x1 = xr[l + 32], x2 = xr[l + 64], x3 = xr[l + 96];
  float ss = x0 * x0 + x1 * x1 + x2 * x2 + x3 * x3;
  for (int off = 1; off < 32; off <<= 1) ss += __shfl_xor(ss, off, 32);
  float inv = rsqrtf(ss / (float)cHD + 1e-6f);
  x0 *= inv * w[l];      x1 *= inv * w[l + 32];
  x2 *= inv * w[l + 64]; x3 *= inv * w[l + 96];

  const float lnTheta = 13.815510557964274f;       // ln(1e6)
  float f0 = __expf(-((float)(2 * l) / 128.f) * lnTheta);
  float f1 = __expf(-((float)(2 * (l + 32)) / 128.f) * lnTheta);
  float c0 = __cosf((float)s * f0), sn0 = __sinf((float)s * f0);
  float c1 = __cosf((float)s * f1), sn1 = __sinf((float)s * f1);
  xr[l]      = x0 * c0 - x2 * sn0;
  xr[l + 64] = x2 * c0 + x0 * sn0;
  xr[l + 32] = x1 * c1 - x3 * sn1;
  xr[l + 96] = x3 * c1 + x1 * sn1;
}

// ---------------- Flash attention, WMMA, 4 waves x 16-query tiles -----------
__global__ __launch_bounds__(128)
void attn_wmma_kernel(const float* __restrict__ q, const float* __restrict__ k,
                      const float* __restrict__ v, float* __restrict__ out)
{
  __shared__ alignas(64) _Float16 qs[64][cHD];     // 16 KB, 4 query tiles
  __shared__ alignas(64) _Float16 ks[32][cHD];     //  8 KB, [key][dim]
  __shared__ alignas(64) _Float16 vt[cHD][32];     //  8 KB, [dim][key]
  __shared__ alignas(64) _Float16 ps[4][16][32];   //  4 KB, per-wave P tile

  const int tid  = threadIdx.x;
  const int lane = tid & 31;
  const int wave = tid >> 5;
  const int hi   = (lane >= 16) ? 1 : 0;
  const int ln   = lane & 15;
  const int q0b  = blockIdx.x * 64;          // block query base
  const int q0w  = q0b + wave * 16;          // this wave's query tile
  const int bh   = blockIdx.y;               // b*NH + h
  const int b    = bh / cNH, h = bh % cNH;
  const int hkv  = h / (cNH / cNKV);
  const float qscale = 0.08838834764831845f; // 1/sqrt(128)

  { // stage Q (64 x 128), f32 -> f16, scaled, contiguous stores
    const int m  = tid >> 1;
    const int c0 = (tid & 1) * 64;
    const float4* qp = (const float4*)(q + (((size_t)b * cS + q0b + m) * cNH + h) * cHD + c0);
#pragma unroll
    for (int i = 0; i < 16; ++i) {
      float4 f = qp[i];
      qs[m][c0 + 4 * i + 0] = (_Float16)(f.x * qscale);
      qs[m][c0 + 4 * i + 1] = (_Float16)(f.y * qscale);
      qs[m][c0 + 4 * i + 2] = (_Float16)(f.z * qscale);
      qs[m][c0 + 4 * i + 3] = (_Float16)(f.w * qscale);
    }
  }
  __syncthreads();

  v16h aq[4];
#pragma unroll
  for (int c = 0; c < 4; ++c)
    aq[c] = ld_frag_row(&qs[wave * 16 + ln][c * 32], hi);

  v8f o[8];
#pragma unroll
  for (int t = 0; t < 8; ++t) o[t] = (v8f){};
  float mrow[8], lrow[8];
#pragma unroll
  for (int r = 0; r < 8; ++r) { mrow[r] = -1e30f; lrow[r] = 0.f; }

  const int jend = q0b + 64;                 // uniform across the block
  for (int j0 = 0; j0 < jend; j0 += 32) {
    { // stage K row-major (32x128): contiguous dim runs, vector LDS stores
      const int jr = tid >> 2;
      const int c0 = (tid & 3) * 32;
      const float4* kp = (const float4*)(k + (((size_t)b * cS + j0 + jr) * cNKV + hkv) * cHD + c0);
#pragma unroll
      for (int i = 0; i < 8; ++i) {
        float4 f = kp[i];
        ks[jr][c0 + 4 * i + 0] = (_Float16)f.x;
        ks[jr][c0 + 4 * i + 1] = (_Float16)f.y;
        ks[jr][c0 + 4 * i + 2] = (_Float16)f.z;
        ks[jr][c0 + 4 * i + 3] = (_Float16)f.w;
      }
    }
    { // stage V transposed vt[dim][key]: one thread per dim packs 8-key runs
      const int d = tid;                       // 0..127
      const float* vcol = v + (((size_t)b * cS + j0) * cNKV + hkv) * cHD + d;
      const size_t rstride = (size_t)cNKV * cHD;
#pragma unroll
      for (int kc = 0; kc < 4; ++kc) {
        union { _Float16 h[8]; v8h v8; } pk;
#pragma unroll
        for (int j = 0; j < 8; ++j)
          pk.h[j] = (_Float16)vcol[(size_t)(kc * 8 + j) * rstride];
        *(v8h*)&vt[d][kc * 8] = pk.v8;
      }
    }
    __syncthreads();

    // scores S[16 x 32] = Q (16x128) @ K^T (128x32)
    v8f s0 = {}, s1 = {};
#pragma unroll
    for (int c = 0; c < 4; ++c) {
      v16h bk0 = ld_frag_contig(&ks[ln][c * 32], hi);
      v16h bk1 = ld_frag_contig(&ks[16 + ln][c * 32], hi);
      s0 = __builtin_amdgcn_wmma_f32_16x16x32_f16(false, aq[c], false, bk0, (short)0, s0, false, false);
      s1 = __builtin_amdgcn_wmma_f32_16x16x32_f16(false, aq[c], false, bk1, (short)0, s1, false, false);
    }

    // causal mask + online softmax (row = 16-lane group in each VGPR)
#pragma unroll
    for (int r = 0; r < 8; ++r) {
      int m = hi ? r + 8 : r;
      float x0 = (j0 + ln      <= q0w + m) ? s0[r] : -1e30f;
      float x1 = (j0 + 16 + ln <= q0w + m) ? s1[r] : -1e30f;
      float tmax = fmaxf(x0, x1);
      for (int off = 1; off < 16; off <<= 1) tmax = fmaxf(tmax, __shfl_xor(tmax, off, 32));
      float mnew  = fmaxf(mrow[r], tmax);
      float alpha = __expf(mrow[r] - mnew);
      float p0 = __expf(x0 - mnew);
      float p1 = __expf(x1 - mnew);
      float rsum = p0 + p1;
      for (int off = 1; off < 16; off <<= 1) rsum += __shfl_xor(rsum, off, 32);
      lrow[r] = lrow[r] * alpha + rsum;
      mrow[r] = mnew;
#pragma unroll
      for (int t = 0; t < 8; ++t) o[t][r] *= alpha;
      ps[wave][m][ln]      = (_Float16)p0;
      ps[wave][m][16 + ln] = (_Float16)p1;
    }

    // O += P (16x32) @ V (32x128)   (ps is per-wave; no block barrier needed)
    v16h ap = ld_frag_row(&ps[wave][ln][0], hi);
#pragma unroll
    for (int t = 0; t < 8; ++t) {
      v16h bv = ld_frag_contig(&vt[t * 16 + ln][0], hi);
      o[t] = __builtin_amdgcn_wmma_f32_16x16x32_f16(false, ap, false, bv, (short)0, o[t], false, false);
    }
    __syncthreads();
  }

#pragma unroll
  for (int r = 0; r < 8; ++r) {
    int m = hi ? r + 8 : r;
    float inv = 1.0f / lrow[r];
#pragma unroll
    for (int t = 0; t < 8; ++t)
      out[(((size_t)b * cS + q0w + m) * cNH + h) * cHD + t * 16 + ln] = o[t][r] * inv;
  }
}

// ---------------- Router: logits, softmax, top-4, combine^T [E,T] -----------
__global__ void router_logits_kernel(const float* __restrict__ xn,
                                     const float* __restrict__ Wr,
                                     float* __restrict__ logits)
{
  int idx = blockIdx.x * blockDim.x + threadIdx.x;   // t*E + e
  if (idx >= cT * cE) return;
  int t = idx / cE, e = idx % cE;
  float acc = 0.f;
  for (int hh = 0; hh < cH; ++hh)
    acc += xn[(size_t)t * cH + hh] * Wr[(size_t)hh * cE + e];
  logits[idx] = acc;
}

__global__ void router_topk_kernel(const float* __restrict__ logits,
                                   float* __restrict__ combineT)
{
  int t = blockIdx.x * blockDim.x + threadIdx.x;
  if (t >= cT) return;
  float p[cE];
  float mx = -1e30f;
  for (int e = 0; e < cE; ++e) { p[e] = logits[(size_t)t * cE + e]; mx = fmaxf(mx, p[e]); }
  float sum = 0.f;
  for (int e = 0; e < cE; ++e) { p[e] = __expf(p[e] - mx); sum += p[e]; }
  float rinv = 1.0f / sum;
  for (int e = 0; e < cE; ++e) p[e] *= rinv;

  float cw[cE];
  bool used[cE];
  for (int e = 0; e < cE; ++e) { cw[e] = 0.f; used[e] = false; }
  float wsum = 0.f; int ids[cTOPK]; float wv[cTOPK];
  for (int kk = 0; kk < cTOPK; ++kk) {
    int best = 0; float bv = -1.f;
    for (int e = 0; e < cE; ++e)
      if (!used[e] && p[e] > bv) { bv = p[e]; best = e; }
    used[best] = true; ids[kk] = best; wv[kk] = bv; wsum += bv;
  }
  float winv = 1.0f / wsum;
  for (int kk = 0; kk < cTOPK; ++kk) cw[ids[kk]] = wv[kk] * winv;
  for (int e = 0; e < cE; ++e) combineT[(size_t)e * cT + t] = cw[e];
}

// ---------------- Elementwise helpers ----------------
__global__ void silu_mul_kernel(const float* __restrict__ g, const float* __restrict__ u,
                                float* __restrict__ act, int n)
{
  int i = blockIdx.x * blockDim.x + threadIdx.x;
  if (i >= n) return;
  float gv = g[i];
  act[i] = gv / (1.f + __expf(-gv)) * u[i];
}

__global__ void copy_kernel(const float* __restrict__ src, float* __restrict__ dst, int n)
{
  int i = blockIdx.x * blockDim.x + threadIdx.x;
  if (i < n) dst[i] = src[i];
}

// ---------------------------------------------------------------------------
extern "C" void kernel_launch(void* const* d_in, const int* in_sizes, int n_in,
                              void* d_out, int out_size, void* d_ws, size_t ws_size,
                              hipStream_t stream)
{
  (void)in_sizes; (void)n_in; (void)out_size; (void)ws_size;
  const float* x        = (const float*)d_in[0];
  const float* ln1_w    = (const float*)d_in[1];
  const float* ln2_w    = (const float*)d_in[2];
  const float* q_norm_w = (const float*)d_in[3];
  const float* k_norm_w = (const float*)d_in[4];
  const float* Wq       = (const float*)d_in[5];
  const float* Wk       = (const float*)d_in[6];
  const float* Wv       = (const float*)d_in[7];
  const float* Wo       = (const float*)d_in[8];
  const float* Wr       = (const float*)d_in[9];
  const float* Wg       = (const float*)d_in[10];
  const float* Wu       = (const float*)d_in[11];
  const float* Wd       = (const float*)d_in[12];
  float* out = (float*)d_out;

  // workspace layout (floats)
  float* ws     = (float*)d_ws;
  float* xn     = ws;                                 // T*H
  float* qb     = xn   + (size_t)cT * cH;             // T*NH*HD
  float* kb     = qb   + (size_t)cT * cNH * cHD;      // T*NKV*HD
  float* vb     = kb   + (size_t)cT * cNKV * cHD;     // T*NKV*HD
  float* attn   = vb   + (size_t)cT * cNKV * cHD;     // T*H
  float* hbuf   = attn + (size_t)cT * cH;             // T*H
  float* xn2    = hbuf + (size_t)cT * cH;             // T*H
  float* logits = xn2  + (size_t)cT * cH;             // T*E
  float* combT  = logits + (size_t)cT * cE;           // E*T
  float* Gb     = combT  + (size_t)cE * cT;           // T*I
  float* Ub     = Gb   + (size_t)cT * cI;             // T*I
  float* actb   = Ub   + (size_t)cT * cI;             // T*I

  // 1) ln1
  rmsnorm_kernel<<<cT, 256, 0, stream>>>(x, ln1_w, xn, cH);

  // 2) QKV projections (WMMA)
  gemm_f16_wmma<<<dim3((cNH * cHD) / GTN, cT / GTM), 256, 0, stream>>>(
      xn, Wq, qb, nullptr, nullptr, cT, cNH * cHD, cH, 0);
  gemm_f16_wmma<<<dim3((cNKV * cHD) / GTN, cT / GTM), 256, 0, stream>>>(
      xn, Wk, kb, nullptr, nullptr, cT, cNKV * cHD, cH, 0);
  gemm_f16_wmma<<<dim3((cNKV * cHD) / GTN, cT / GTM), 256, 0, stream>>>(
      xn, Wv, vb, nullptr, nullptr, cT, cNKV * cHD, cH, 0);

  // 3) per-head QK norm + RoPE
  qknorm_rope_kernel<<<cT * cNH, 32, 0, stream>>>(qb, q_norm_w, cNH);
  qknorm_rope_kernel<<<cT * cNKV, 32, 0, stream>>>(kb, k_norm_w, cNKV);

  // 4) causal flash attention (WMMA, 4 query tiles per block)
  attn_wmma_kernel<<<dim3(cS / 64, cB * cNH), 128, 0, stream>>>(qb, kb, vb, attn);

  // 5) O-proj + residual
  gemm_f16_wmma<<<dim3(cH / GTN, cT / GTM), 256, 0, stream>>>(
      attn, Wo, hbuf, x, nullptr, cT, cH, cH, 1);

  // 6) ln2
  rmsnorm_kernel<<<cT, 256, 0, stream>>>(hbuf, ln2_w, xn2, cH);

  // 7) router
  router_logits_kernel<<<(cT * cE + 255) / 256, 256, 0, stream>>>(xn2, Wr, logits);
  router_topk_kernel<<<(cT + 255) / 256, 256, 0, stream>>>(logits, combT);

  // 8) out = h ; dense-MoE accumulates combine-weighted expert outputs
  copy_kernel<<<(cT * cH + 255) / 256, 256, 0, stream>>>(hbuf, out, cT * cH);

  for (int e = 0; e < cE; ++e) {
    const float* wg = Wg + (size_t)e * cH * cI;
    const float* wu = Wu + (size_t)e * cH * cI;
    const float* wd = Wd + (size_t)e * cI * cH;
    gemm_f16_wmma<<<dim3(cI / GTN, cT / GTM), 256, 0, stream>>>(
        xn2, wg, Gb, nullptr, nullptr, cT, cI, cH, 0);
    gemm_f16_wmma<<<dim3(cI / GTN, cT / GTM), 256, 0, stream>>>(
        xn2, wu, Ub, nullptr, nullptr, cT, cI, cH, 0);
    silu_mul_kernel<<<(cT * cI + 255) / 256, 256, 0, stream>>>(Gb, Ub, actb, cT * cI);
    gemm_f16_wmma<<<dim3(cH / GTN, cT / GTM), 256, 0, stream>>>(
        actb, wd, out, nullptr, combT + (size_t)e * cT, cT, cH, cI, 2);
  }
}
